// HymbaBlock_75746043232516
// MI455X (gfx1250) — matrix operand order repaired and checked by reading, not compiled, not run on hardware
//
#include <hip/hip_runtime.h>

typedef _Float16 v16h __attribute__((ext_vector_type(16)));
typedef _Float16 v8h  __attribute__((ext_vector_type(8)));
typedef float    v8f  __attribute__((ext_vector_type(8)));

__device__ __forceinline__ v16h cat8(v8h lo, v8h hi) {
  return __builtin_shufflevector(lo, hi, 0,1,2,3,4,5,6,7,8,9,10,11,12,13,14,15);
}

__device__ __forceinline__ v8f wmma_f16(v16h a, v16h b, v8f c) {
  // D = A(16x32) * B(32x16) + C, fp32 accumulate
  return __builtin_amdgcn_wmma_f32_16x16x32_f16(false, a, false, b, (short)0, c, false, false);
}

__device__ __forceinline__ v8f vzero8() {
  v8f z = {0.f,0.f,0.f,0.f,0.f,0.f,0.f,0.f};
  return z;
}

// A-fragment (16x32 f16): lane m = lane&15, half = lane>>4.
// elements 0..7  -> k = half*16 + e        (contiguous 8 halfs)
// elements 8..15 -> k = half*16 + 8 + e    (contiguous 8 halfs)
__device__ __forceinline__ v16h load_afrag(const _Float16* __restrict__ row, int k0, int half) {
  v8h lo = *reinterpret_cast<const v8h*>(row + k0 + half * 16);
  v8h hi = *reinterpret_cast<const v8h*>(row + k0 + half * 16 + 8);
  return cat8(lo, hi);
}

// B-fragment (32x16 f16): lane n = lane&15, half = lane>>4; element e -> k = half*16 + e
// For C = A @ B^T with B stored row-major (N x K), lane reads 16 contiguous halfs of row n.
__device__ __forceinline__ v16h load_bfrag(const _Float16* __restrict__ row, int k0, int half) {
  return *reinterpret_cast<const v16h*>(row + k0 + half * 16);
}

// ---------------------------------------------------------------------------
// fp32 -> f16 conversion
__global__ void cvt_f32_f16(const float* __restrict__ s, _Float16* __restrict__ d, int n) {
  int i = blockIdx.x * 256 + threadIdx.x;
  if (i < n) d[i] = (_Float16)s[i];
}

// Pack Wq (768x768), Wk (192x768), Wv (192x768) -> Wqkv (1152x768) f16
__global__ void pack_wqkv(const float* __restrict__ Wq, const float* __restrict__ Wk,
                          const float* __restrict__ Wv, _Float16* __restrict__ W) {
  int i = blockIdx.x * 256 + threadIdx.x;
  if (i >= 1152 * 768) return;
  int r = i / 768, c = i - r * 768;
  float v;
  if (r < 768)      v = Wq[r * 768 + c];
  else if (r < 960) v = Wk[(r - 768) * 768 + c];
  else              v = Wv[(r - 960) * 768 + c];
  W[i] = (_Float16)v;
}

// ---------------------------------------------------------------------------
// QKV GEMM: Xh (4096 x 768) @ Wqkv^T (768 x 1152) -> scatter into
//  Qh (2,12,2048,64), Kh (2,3,2048,64), Vt (2,3,64,2048) all f16
__global__ __launch_bounds__(128) void gemm_qkv_kernel(
    const _Float16* __restrict__ X, const _Float16* __restrict__ Wqkv,
    _Float16* __restrict__ Qh, _Float16* __restrict__ Kh, _Float16* __restrict__ Vt) {
  const int K = 768;
  int lane = threadIdx.x & 31;
  int wave = threadIdx.x >> 5;
  int nl = lane & 15, half = lane >> 4;
  int m0 = blockIdx.x * 64;
  int n0 = blockIdx.y * 64 + wave * 16;

  v8f acc[4];
#pragma unroll
  for (int mt = 0; mt < 4; ++mt) acc[mt] = vzero8();

  const _Float16* brow = Wqkv + (size_t)(n0 + nl) * K;
  for (int k0 = 0; k0 < K; k0 += 32) {
    if (k0 + 32 < K) {
      __builtin_prefetch(brow + k0 + 32 + half * 16, 0, 3);
    }
    v16h bf = load_bfrag(brow, k0, half);
#pragma unroll
    for (int mt = 0; mt < 4; ++mt) {
      v16h af = load_afrag(X + (size_t)(m0 + mt * 16 + nl) * K, k0, half);
      acc[mt] = wmma_f16(af, bf, acc[mt]);
    }
  }

#pragma unroll
  for (int mt = 0; mt < 4; ++mt) {
#pragma unroll
    for (int r = 0; r < 8; ++r) {
      int row = m0 + mt * 16 + r + half * 8;   // token row: b*2048 + t
      int col = n0 + nl;                        // feature column
      int b = row >> 11, t = row & 2047;
      _Float16 hv = (_Float16)acc[mt][r];
      if (col < 768) {
        int hh = col >> 6, d = col & 63;
        Qh[(((size_t)b * 12 + hh) * 2048 + t) * 64 + d] = hv;
      } else if (col < 960) {
        int c2 = col - 768; int kvh = c2 >> 6, d = c2 & 63;
        Kh[(((size_t)b * 3 + kvh) * 2048 + t) * 64 + d] = hv;
      } else {
        int c2 = col - 960; int kvh = c2 >> 6, d = c2 & 63;
        Vt[(((size_t)b * 3 + kvh) * 64 + d) * 2048 + t] = hv;   // transposed V
      }
    }
  }
}

// ---------------------------------------------------------------------------
// Rotary: buf layout (2, nheads, 2048, 64); pair (i, i+32), i in [0,32)
__global__ void rotary_kernel(_Float16* __restrict__ buf, int total) {
  int idx = blockIdx.x * 256 + threadIdx.x;
  if (idx >= total) return;
  int i = idx & 31;
  int rest = idx >> 5;
  int t = rest & 2047;          // T = 2048
  int bh = rest >> 11;
  _Float16* p = buf + ((size_t)bh * 2048 + t) * 64;
  float invf = __expf(-(float)(2 * i) * (1.0f / 64.0f) * 9.210340371976184f); // 10000^(-2i/64)
  float fr = (float)t * invf;
  float s, c;
  __sincosf(fr, &s, &c);
  float x1 = (float)p[i], x2 = (float)p[32 + i];
  p[i]      = (_Float16)(x1 * c - x2 * s);
  p[32 + i] = (_Float16)(x1 * s + x2 * c);
}

// ---------------------------------------------------------------------------
// Sliding-window flash attention. One wave per (b, h, 16-query tile).
__global__ __launch_bounds__(32) void attn_kernel(
    const _Float16* __restrict__ Qh, const _Float16* __restrict__ Kh,
    const _Float16* __restrict__ Vt, _Float16* __restrict__ Xo) {
  const int T = 2048, D = 64;
  const float SCALE = 0.21650635094610965f;  // (64*4/12)^-0.5
  __shared__ __align__(32) _Float16 Plds[16 * 32];

  int qt = blockIdx.x, h = blockIdx.y, b = blockIdx.z;
  int kvh = h >> 2;                 // jnp.repeat(k, 4, axis=1): head h uses kv head h/4
  int lane = threadIdx.x;
  int nl = lane & 15, half = lane >> 4;
  int i0 = qt * 16;

  const _Float16* Qp = Qh + (((size_t)b * 12 + h) * T) * D;
  const _Float16* Kp = Kh + (((size_t)b * 3 + kvh) * T) * D;
  const _Float16* Vp = Vt + (((size_t)b * 3 + kvh) * D) * T;

  // Q fragments for this tile (rows i0+nl), K' = 64 -> two 32-wide steps
  v16h aq0 = load_afrag(Qp + (size_t)(i0 + nl) * D, 0, half);
  v16h aq1 = load_afrag(Qp + (size_t)(i0 + nl) * D, 32, half);

  v8f o0 = vzero8(), o1 = vzero8(), o2 = vzero8(), o3 = vzero8();
  float mstat[8], lstat[8];
  int st_lane[8];
#pragma unroll
  for (int r = 0; r < 8; ++r) {
    mstat[r] = -INFINITY;
    lstat[r] = 0.f;
    int ir = i0 + r + half * 8;
    int st = ir - 512;
    st = st < 0 ? 0 : (st > 1024 ? 1024 : st);
    st_lane[r] = st;
  }

  int smin = i0 - 512;      smin = smin < 0 ? 0 : (smin > 1024 ? 1024 : smin);
  int smax = i0 + 15 - 512; smax = smax < 0 ? 0 : (smax > 1024 ? 1024 : smax);
  int kk0 = smin & ~31;
  int kkend = smax + 1024;

  for (int kk = kk0; kk < kkend; kk += 32) {
    // S(16x32) = Q(16x64) @ K[kk..kk+31]^T  (two 16x16 N-tiles)
    v8f s0 = vzero8(), s1 = vzero8();
    {
      const _Float16* kr0 = Kp + (size_t)(kk + nl) * D;
      v16h b00 = *reinterpret_cast<const v16h*>(kr0 + half * 16);
      v16h b01 = *reinterpret_cast<const v16h*>(kr0 + 32 + half * 16);
      s0 = wmma_f16(aq0, b00, s0);
      s0 = wmma_f16(aq1, b01, s0);
      const _Float16* kr1 = Kp + (size_t)(kk + 16 + nl) * D;
      v16h b10 = *reinterpret_cast<const v16h*>(kr1 + half * 16);
      v16h b11 = *reinterpret_cast<const v16h*>(kr1 + 32 + half * 16);
      s1 = wmma_f16(aq0, b10, s1);
      s1 = wmma_f16(aq1, b11, s1);
    }

    int j0 = kk + nl, j1 = kk + 16 + nl;
#pragma unroll
    for (int r = 0; r < 8; ++r) {
      int st = st_lane[r];
      float v0 = (j0 >= st && j0 < st + 1024) ? s0[r] * SCALE : -INFINITY;
      float v1 = (j1 >= st && j1 < st + 1024) ? s1[r] * SCALE : -INFINITY;
      // row max over 32 keys (reduce over 16-lane group; D-layout: lane&15 = column)
      float mx = fmaxf(v0, v1);
      mx = fmaxf(mx, __shfl_xor(mx, 1));
      mx = fmaxf(mx, __shfl_xor(mx, 2));
      mx = fmaxf(mx, __shfl_xor(mx, 4));
      mx = fmaxf(mx, __shfl_xor(mx, 8));
      float mtot = fmaxf(mstat[r], mx);
      float me = fmaxf(mtot, -1e30f);     // NaN-safe subtrahend for all-masked rows
      float so = __expf(mstat[r] - me);
      float e0 = __expf(v0 - me);
      float e1 = __expf(v1 - me);
      float rs = e0 + e1;
      rs += __shfl_xor(rs, 1);
      rs += __shfl_xor(rs, 2);
      rs += __shfl_xor(rs, 4);
      rs += __shfl_xor(rs, 8);
      lstat[r] = lstat[r] * so + rs;
      mstat[r] = mtot;
      o0[r] = o0[r] * so; o1[r] = o1[r] * so;
      o2[r] = o2[r] * so; o3[r] = o3[r] * so;
      int prow = r + half * 8;
      Plds[prow * 32 + nl] = (_Float16)e0;
      Plds[prow * 32 + 16 + nl] = (_Float16)e1;
    }
    __syncthreads();
    // Re-read P as an A-fragment (transpose D-layout -> A-layout via LDS)
    v8h plo = *reinterpret_cast<const v8h*>(&Plds[nl * 32 + half * 16]);
    v8h phi = *reinterpret_cast<const v8h*>(&Plds[nl * 32 + half * 16 + 8]);
    v16h pa = cat8(plo, phi);
    __syncthreads();

    // O(16x64) += P(16x32) @ V(32x64), V stored transposed -> contiguous B rows
    {
      v16h bv0 = *reinterpret_cast<const v16h*>(Vp + (size_t)(0 * 16 + nl) * T + kk + half * 16);
      o0 = wmma_f16(pa, bv0, o0);
      v16h bv1 = *reinterpret_cast<const v16h*>(Vp + (size_t)(1 * 16 + nl) * T + kk + half * 16);
      o1 = wmma_f16(pa, bv1, o1);
      v16h bv2 = *reinterpret_cast<const v16h*>(Vp + (size_t)(2 * 16 + nl) * T + kk + half * 16);
      o2 = wmma_f16(pa, bv2, o2);
      v16h bv3 = *reinterpret_cast<const v16h*>(Vp + (size_t)(3 * 16 + nl) * T + kk + half * 16);
      o3 = wmma_f16(pa, bv3, o3);
    }
  }

#pragma unroll
  for (int r = 0; r < 8; ++r) {
    int ir = i0 + r + half * 8;
    float inv = 1.0f / lstat[r];
    size_t basep = ((size_t)b * T + ir) * 768 + (size_t)h * 64;
    Xo[basep + 0 * 16 + nl] = (_Float16)(o0[r] * inv);
    Xo[basep + 1 * 16 + nl] = (_Float16)(o1[r] * inv);
    Xo[basep + 2 * 16 + nl] = (_Float16)(o2[r] * inv);
    Xo[basep + 3 * 16 + nl] = (_Float16)(o3[r] * inv);
  }
}

// ---------------------------------------------------------------------------
// Output GEMM: Xo (4096 x 768) @ Wo^T (768 x 768) -> out fp32 (4096 x 768)
__global__ __launch_bounds__(128) void gemm_out_kernel(
    const _Float16* __restrict__ X, const _Float16* __restrict__ Wo16,
    float* __restrict__ out) {
  const int K = 768;
  int lane = threadIdx.x & 31;
  int wave = threadIdx.x >> 5;
  int nl = lane & 15, half = lane >> 4;
  int m0 = blockIdx.x * 64;
  int n0 = blockIdx.y * 64 + wave * 16;

  v8f acc[4];
#pragma unroll
  for (int mt = 0; mt < 4; ++mt) acc[mt] = vzero8();

  const _Float16* brow = Wo16 + (size_t)(n0 + nl) * K;
  for (int k0 = 0; k0 < K; k0 += 32) {
    if (k0 + 32 < K) {
      __builtin_prefetch(brow + k0 + 32 + half * 16, 0, 3);
    }
    v16h bf = load_bfrag(brow, k0, half);
#pragma unroll
    for (int mt = 0; mt < 4; ++mt) {
      v16h af = load_afrag(X + (size_t)(m0 + mt * 16 + nl) * K, k0, half);
      acc[mt] = wmma_f16(af, bf, acc[mt]);
    }
  }

#pragma unroll
  for (int mt = 0; mt < 4; ++mt) {
#pragma unroll
    for (int r = 0; r < 8; ++r) {
      int row = m0 + mt * 16 + r + half * 8;
      int col = n0 + nl;
      out[(size_t)row * 768 + col] = acc[mt][r];
    }
  }
}

// ---------------------------------------------------------------------------
extern "C" void kernel_launch(void* const* d_in, const int* in_sizes, int n_in,
                              void* d_out, int out_size, void* d_ws, size_t ws_size,
                              hipStream_t stream) {
  (void)in_sizes; (void)n_in; (void)out_size; (void)ws_size;
  const float* hs = (const float*)d_in[0];
  const float* Wq = (const float*)d_in[1];
  const float* Wk = (const float*)d_in[2];
  const float* Wv = (const float*)d_in[3];
  const float* Wo = (const float*)d_in[4];
  float* out = (float*)d_out;

  char* base = (char*)d_ws;
  size_t off = 0;
  auto take = [&](size_t bytes) -> void* {
    void* p = base + off;
    off += (bytes + 255) & ~(size_t)255;
    return p;
  };
  _Float16* Xh   = (_Float16*)take((size_t)4096 * 768 * 2);       // hidden f16
  _Float16* Wqkv = (_Float16*)take((size_t)1152 * 768 * 2);       // packed QKV weights
  _Float16* Wo16 = (_Float16*)take((size_t)768 * 768 * 2);        // Wo f16
  _Float16* Qh   = (_Float16*)take((size_t)2 * 12 * 2048 * 64 * 2);
  _Float16* Kh   = (_Float16*)take((size_t)2 * 3 * 2048 * 64 * 2);
  _Float16* Vt   = (_Float16*)take((size_t)2 * 3 * 2048 * 64 * 2); // V transposed
  _Float16* Xo   = (_Float16*)take((size_t)4096 * 768 * 2);        // attn output

  const int nX  = 2 * 2048 * 768;   // 3145728
  const int nWc = 1152 * 768;       // 884736
  const int nWo = 768 * 768;        // 589824

  cvt_f32_f16<<<(nX + 255) / 256, 256, 0, stream>>>(hs, Xh, nX);
  pack_wqkv<<<(nWc + 255) / 256, 256, 0, stream>>>(Wq, Wk, Wv, Wqkv);
  cvt_f32_f16<<<(nWo + 255) / 256, 256, 0, stream>>>(Wo, Wo16, nWo);

  gemm_qkv_kernel<<<dim3(64, 18), 128, 0, stream>>>(Xh, Wqkv, Qh, Kh, Vt);

  const int nQrot = 2 * 12 * 2048 * 32;  // 1572864
  const int nKrot = 2 * 3 * 2048 * 32;   // 393216
  rotary_kernel<<<(nQrot + 255) / 256, 256, 0, stream>>>(Qh, nQrot);
  rotary_kernel<<<(nKrot + 255) / 256, 256, 0, stream>>>(Kh, nKrot);

  attn_kernel<<<dim3(128, 12, 2), 32, 0, stream>>>(Qh, Kh, Vt, Xo);

  gemm_out_kernel<<<dim3(64, 12), 128, 0, stream>>>(Xo, Wo16, out);
}
